// Net_7464653160908
// MI455X (gfx1250) — compile-verified, matrix operand order checked by reading, not run on hardware
//
#include <hip/hip_runtime.h>
#include <hip/hip_bf16.h>
#include <math.h>

typedef __attribute__((ext_vector_type(16))) _Float16 v16h;
typedef __attribute__((ext_vector_type(8)))  _Float16 v8h;
typedef __attribute__((ext_vector_type(2)))  _Float16 v2h;
typedef __attribute__((ext_vector_type(8)))  float    v8f;
typedef __attribute__((ext_vector_type(4)))  unsigned int v4u;
typedef __attribute__((ext_vector_type(8)))  int v8i;
typedef __attribute__((ext_vector_type(4)))  int v4i;

#define BATCH   8192
#define NCOEF   13          // GRID_N + K_ORDER
#define FEAT    14          // 13 basis + mish
#define K1PAD   768         // 49*14=686 padded to mult of 256 (chunking)
#define K23     3584        // 256*14, mult of 256
#define CHUNK_KB 8          // kb-blocks (32 K each) staged per TDM chunk
#define FRAG_HALFS 512      // one 32K x 16N fragment = 32 lanes * 16 halfs = 1KB

// ---------------------------------------------------------------------------
// B-spline basis (uniform knots, degree 3 Cox-de Boor) + mish, fp32 internal.
// knots t_j = -3.2 + 0.4*j, j = 0..16  (matches reference exactly)
// ---------------------------------------------------------------------------
__device__ __forceinline__ void kan_features(float x, _Float16* out14) {
    const float t0 = -3.2f, hs = 0.4f;
    float B[16];
#pragma unroll
    for (int j = 0; j < 16; ++j) {
        float tj = t0 + hs * (float)j;
        B[j] = (x >= tj && x < tj + hs) ? 1.0f : 0.0f;
    }
#pragma unroll
    for (int d = 1; d <= 3; ++d) {
        float inv = 1.0f / (hs * (float)d);
#pragma unroll
        for (int j = 0; j < 16 - d; ++j) {
            float tj = t0 + hs * (float)j;
            B[j] = (x - tj) * inv * B[j] + (tj + hs * (float)(d + 1) - x) * inv * B[j + 1];
        }
    }
#pragma unroll
    for (int j = 0; j < NCOEF; ++j) out14[j] = (_Float16)B[j];
    float sp = (x > 15.0f) ? x : log1pf(expf(x));
    out14[NCOEF] = (_Float16)(x * tanhf(sp));
}

// ---------------------------------------------------------------------------
// TDM: 2D tile copy global->LDS.  tile_w0 words per strip, ntile1 strips,
// strips strided by stride_w words.  Descriptor per CDNA5 ISA.
// ---------------------------------------------------------------------------
__device__ __forceinline__ void tdm_load_2d(unsigned lds_addr, const void* gptr,
                                            unsigned tile_w0, unsigned ntile1,
                                            unsigned stride_w) {
    unsigned long long ga = (unsigned long long)(size_t)gptr;
    v4u g0;
    g0[0] = 1u;                                               // count=1 (valid), user mode
    g0[1] = lds_addr;                                         // lds_addr (bytes)
    g0[2] = (unsigned)ga;                                     // global_addr[31:0]
    g0[3] = (unsigned)((ga >> 32) & 0x01FFFFFFu) | (2u << 30);// addr[56:32] | type=2
    const unsigned TD0 = stride_w;                            // tensor_dim0 >= tile span
    const unsigned TD1 = 0x7FFFFFFFu;                         // no OOB in dim1
    v8i g1;
    g1[0] = (int)(2u << 16);                                  // data_size = 4 bytes
    g1[1] = (int)((TD0 & 0xFFFFu) << 16);                     // atomic_barrier=0 | dim0 lo
    g1[2] = (int)(((TD0 >> 16) & 0xFFFFu) | ((TD1 & 0xFFFFu) << 16));
    g1[3] = (int)(((TD1 >> 16) & 0xFFFFu) | ((tile_w0 & 0xFFFFu) << 16)); // tile_dim0
    g1[4] = (int)(ntile1 & 0xFFFFu);                          // tile_dim1 | tile_dim2=0
    g1[5] = (int)stride_w;                                    // tensor_dim0_stride[31:0]
    g1[6] = 0;                                                // stride hi | dim1_stride lo
    g1[7] = 0;
    v4i z4 = {0, 0, 0, 0};
#if defined(__clang_major__) && (__clang_major__ >= 23)
    v8i z8 = {0, 0, 0, 0, 0, 0, 0, 0};
    __builtin_amdgcn_tensor_load_to_lds(g0, g1, z4, z4, z8, 0);
#else
    __builtin_amdgcn_tensor_load_to_lds(g0, g1, z4, z4, 0);
#endif
}

// ---------------------------------------------------------------------------
// Kernel 1: fused 4x4 avg-pool + layer-1 feature build (f16), pads K to 768.
// ---------------------------------------------------------------------------
__global__ __launch_bounds__(256) void pool_feat1(const float* __restrict__ x,
                                                  _Float16* __restrict__ F1) {
    int tid = blockIdx.x * 256 + threadIdx.x;
    if (tid >= BATCH * 49) return;
    int b = tid / 49, i = tid % 49;
    int oy = i / 7, ox = i % 7;
    const float* src = x + (size_t)b * 784 + (size_t)(oy * 4) * 28 + ox * 4;
    float s = 0.0f;
#pragma unroll
    for (int r = 0; r < 4; ++r)
#pragma unroll
        for (int c = 0; c < 4; ++c) s += src[r * 28 + c];
    float h = s * (1.0f / 16.0f);

    _Float16 f[FEAT];
    kan_features(h, f);
    _Float16* dst = F1 + (size_t)b * K1PAD + i * FEAT;
#pragma unroll
    for (int j = 0; j < 7; ++j) {
        v2h p; p[0] = f[2 * j]; p[1] = f[2 * j + 1];
        *(v2h*)(dst + 2 * j) = p;
    }
    if (i == 48) {  // zero-pad cols 686..767  (82 halfs)
        _Float16* pz = F1 + (size_t)b * K1PAD + 49 * FEAT;
        v2h z; z[0] = (_Float16)0.0f; z[1] = (_Float16)0.0f;
#pragma unroll
        for (int j = 0; j < 41; ++j) *(v2h*)(pz + 2 * j) = z;
    }
}

// ---------------------------------------------------------------------------
// Feature build for layers 2/3: H (B,256) fp32 -> F (B,3584) f16
// ---------------------------------------------------------------------------
__global__ __launch_bounds__(256) void kan_feat(const float* __restrict__ H,
                                                _Float16* __restrict__ F) {
    int tid = blockIdx.x * 256 + threadIdx.x;
    if (tid >= BATCH * 256) return;
    float x = H[tid];
    _Float16 f[FEAT];
    kan_features(x, f);
    int b = tid >> 8, i = tid & 255;
    _Float16* dst = F + (size_t)b * K23 + i * FEAT;
#pragma unroll
    for (int j = 0; j < 7; ++j) {
        v2h p; p[0] = f[2 * j]; p[1] = f[2 * j + 1];
        *(v2h*)(dst + 2 * j) = p;
    }
}

// ---------------------------------------------------------------------------
// Weight prep -> fragment-ordered f16 B operand, nb-major strips:
//   frag f = nb*KB + kb; within frag, lane l element e:
//   K = kb*32 + (e>>3)*16 + (l>>4)*8 + (e&7);  n = nb*16 + (l&15)
// ---------------------------------------------------------------------------
__global__ __launch_bounds__(256) void prep_w(const float* __restrict__ coef,
                                              const float* __restrict__ sb,
                                              const float* __restrict__ sp,
                                              _Float16* __restrict__ Wf,
                                              int In, int Nout, int KBtot,
                                              int nFrag) {
    int tid = blockIdx.x * 256 + threadIdx.x;
    int f = tid >> 5, lane = tid & 31;
    if (f >= nFrag) return;
    int nb = f / KBtot, kb = f % KBtot;
    int n = nb * 16 + (lane & 15);
    int Ksrc = In * FEAT;
    _Float16 vals[16];
#pragma unroll
    for (int e = 0; e < 16; ++e) {
        int k = kb * 32 + ((e >> 3) << 4) + ((lane >> 4) << 3) + (e & 7);
        float v = 0.0f;
        if (k < Ksrc && n < Nout) {
            int i = k / FEAT, r = k % FEAT;
            v = (r < NCOEF)
                ? coef[((size_t)i * Nout + n) * NCOEF + r] * sp[(size_t)i * Nout + n]
                : sb[(size_t)i * Nout + n];
        }
        vals[e] = (_Float16)v;
    }
    _Float16* dst = Wf + ((size_t)f * 32 + lane) * 16;
#pragma unroll
    for (int e = 0; e < 16; ++e) dst[e] = vals[e];
}

// ---------------------------------------------------------------------------
// WMMA GEMM with TDM-staged, double-buffered B in LDS.
// Block = 8 waves; wave = 16M x (WN*16)N; B chunk = WN strips x CHUNK_KB frags.
// ---------------------------------------------------------------------------
template <int WN>
__global__ __launch_bounds__(256) void kan_gemm(const _Float16* __restrict__ F,
                                                const _Float16* __restrict__ Wf,
                                                const float* __restrict__ bias,
                                                float* __restrict__ H,
                                                int K, int N, int biasN) {
    constexpr int CH = CHUNK_KB;
    constexpr unsigned BUF_HALFS = (unsigned)WN * CH * FRAG_HALFS;
    __shared__ _Float16 smem[2 * WN * CH * FRAG_HALFS];

    const int lane = threadIdx.x & 31;
    const int wave = threadIdx.x >> 5;
    const int mtile = blockIdx.x * 8 + wave;
    const int nb0 = blockIdx.y * WN;
    const int KB = K >> 5;
    const int C = KB / CH;

    const int m = mtile * 16 + (lane & 15);
    const _Float16* arow = F + (size_t)m * K + ((lane >> 4) << 3);

    v8f acc[WN];
#pragma unroll
    for (int t = 0; t < WN; ++t)
#pragma unroll
        for (int j = 0; j < 8; ++j) acc[t][j] = 0.0f;

    const unsigned lds0 = (unsigned)(size_t)(void*)&smem[0];          // LDS byte offset
    const _Float16* wbase = Wf + (size_t)nb0 * KB * FRAG_HALFS;       // first strip
    const unsigned tileW = CH * 256;                                  // words per strip
    const unsigned strideW = (unsigned)KB * 256;                      // words strip->strip

    if (wave == 0) tdm_load_2d(lds0, wbase, tileW, WN, strideW);      // chunk 0 -> buf 0

    // software-pipelined A fragment (one kb ahead)
    v16h a_cur;
    {
        v8h lo = *(const v8h*)(arow);
        v8h hi = *(const v8h*)(arow + 16);
#pragma unroll
        for (int j = 0; j < 8; ++j) { a_cur[j] = lo[j]; a_cur[j + 8] = hi[j]; }
    }

    for (int c = 0; c < C; ++c) {
        if (wave == 0) __builtin_amdgcn_s_wait_tensorcnt(0);
        __syncthreads();                                              // buf[c&1] ready
        if (wave == 0 && c + 1 < C) {
            tdm_load_2d(lds0 + ((unsigned)((c + 1) & 1)) * BUF_HALFS * 2,
                        wbase + (size_t)(c + 1) * CH * FRAG_HALFS,
                        tileW, WN, strideW);
        }
        const _Float16* bbase = smem + (size_t)(c & 1) * BUF_HALFS;
#pragma unroll
        for (int kc = 0; kc < CH; ++kc) {
            int kb = c * CH + kc;
            v16h a_nxt = a_cur;
            if (kb + 1 < KB) {
                const _Float16* p = arow + (size_t)(kb + 1) * 32;
                v8h lo = *(const v8h*)(p);
                v8h hi = *(const v8h*)(p + 16);
#pragma unroll
                for (int j = 0; j < 8; ++j) { a_nxt[j] = lo[j]; a_nxt[j + 8] = hi[j]; }
            }
#pragma unroll
            for (int t = 0; t < WN; ++t) {
                v16h b = *(const v16h*)(bbase + ((size_t)(t * CH + kc) * 32 + lane) * 16);
                acc[t] = __builtin_amdgcn_wmma_f32_16x16x32_f16(
                    false, a_cur, false, b, (short)0, acc[t], false, false);
            }
            a_cur = a_nxt;
        }
        __syncthreads();                                              // done reading buf[c&1]
    }

    // Epilogue: D VGPR r -> row (r + 8*(lane>>4)), col = lane&15 within n-tile.
    int rbase = mtile * 16 + ((lane >> 4) << 3);
    int cl = lane & 15;
#pragma unroll
    for (int t = 0; t < WN; ++t) {
        int col = (nb0 + t) * 16 + cl;
        float bv = (col < biasN) ? bias[col] : 0.0f;
#pragma unroll
        for (int r = 0; r < 8; ++r)
            H[(size_t)(rbase + r) * N + col] = acc[t][r] + bv;
    }
}

// ---------------------------------------------------------------------------
// log_softmax over 10 classes (H3 padded to 16 cols)
// ---------------------------------------------------------------------------
__global__ __launch_bounds__(256) void logsoftmax10(const float* __restrict__ H3,
                                                    float* __restrict__ out) {
    int b = blockIdx.x * 256 + threadIdx.x;
    if (b >= BATCH) return;
    const float* r = H3 + (size_t)b * 16;
    float v[10], mx = -3.4e38f;
#pragma unroll
    for (int j = 0; j < 10; ++j) { v[j] = r[j]; mx = fmaxf(mx, v[j]); }
    float s = 0.0f;
#pragma unroll
    for (int j = 0; j < 10; ++j) s += expf(v[j] - mx);
    float lse = mx + logf(s);
#pragma unroll
    for (int j = 0; j < 10; ++j) out[(size_t)b * 10 + j] = v[j] - lse;
}

// ---------------------------------------------------------------------------
extern "C" void kernel_launch(void* const* d_in, const int* in_sizes, int n_in,
                              void* d_out, int out_size, void* d_ws, size_t ws_size,
                              hipStream_t stream) {
    const float* x     = (const float*)d_in[0];
    const float* coef1 = (const float*)d_in[1];
    const float* sb1   = (const float*)d_in[2];
    const float* sp1   = (const float*)d_in[3];
    const float* b1    = (const float*)d_in[4];
    const float* coef2 = (const float*)d_in[5];
    const float* sb2   = (const float*)d_in[6];
    const float* sp2   = (const float*)d_in[7];
    const float* b2    = (const float*)d_in[8];
    const float* coef3 = (const float*)d_in[9];
    const float* sb3   = (const float*)d_in[10];
    const float* sp3   = (const float*)d_in[11];
    const float* b3    = (const float*)d_in[12];
    float* out = (float*)d_out;

    char* ws = (char*)d_ws;
    size_t off = 0;
    auto alloc = [&](size_t bytes) { void* p = ws + off; off += (bytes + 255) & ~(size_t)255; return p; };
    _Float16* F1  = (_Float16*)alloc((size_t)BATCH * K1PAD * 2);   // 12.6 MB
    _Float16* W1f = (_Float16*)alloc((size_t)K1PAD * 256 * 2);     // 0.39 MB
    float*    H1  = (float*)   alloc((size_t)BATCH * 256 * 4);     // 8.4 MB
    _Float16* F23 = (_Float16*)alloc((size_t)BATCH * K23 * 2);     // 58.7 MB (reused L2/L3)
    _Float16* W2f = (_Float16*)alloc((size_t)K23 * 256 * 2);       // 1.8 MB
    float*    H2  = (float*)   alloc((size_t)BATCH * 256 * 4);     // 8.4 MB
    _Float16* W3f = (_Float16*)alloc((size_t)K23 * 16 * 2);        // 0.11 MB
    float*    H3  = (float*)   alloc((size_t)BATCH * 16 * 4);      // 0.5 MB
    (void)ws_size; (void)n_in; (void)in_sizes; (void)out_size;

    // weight prep (nb-major fragment strips)
    { int kb = K1PAD / 32, nf = kb * (256 / 16);
      prep_w<<<(nf * 32 + 255) / 256, 256, 0, stream>>>(coef1, sb1, sp1, W1f, 49, 256, kb, nf); }
    { int kb = K23 / 32, nf = kb * (256 / 16);
      prep_w<<<(nf * 32 + 255) / 256, 256, 0, stream>>>(coef2, sb2, sp2, W2f, 256, 256, kb, nf); }
    { int kb = K23 / 32, nf = kb * (16 / 16);
      prep_w<<<(nf * 32 + 255) / 256, 256, 0, stream>>>(coef3, sb3, sp3, W3f, 256, 10, kb, nf); }

    // pool + layer-1 features
    pool_feat1<<<(BATCH * 49 + 255) / 256, 256, 0, stream>>>(x, F1);

    // layer 1: (8192,768) x (768,256)
    kan_gemm<4><<<dim3(BATCH / 128, 256 / 64), 256, 0, stream>>>(F1, W1f, b1, H1, K1PAD, 256, 256);

    // layer 2: features + (8192,3584) x (3584,256)
    kan_feat<<<(BATCH * 256 + 255) / 256, 256, 0, stream>>>(H1, F23);
    kan_gemm<4><<<dim3(BATCH / 128, 256 / 64), 256, 0, stream>>>(F23, W2f, b2, H2, K23, 256, 256);

    // layer 3: features + (8192,3584) x (3584,16)
    kan_feat<<<(BATCH * 256 + 255) / 256, 256, 0, stream>>>(H2, F23);
    kan_gemm<1><<<dim3(BATCH / 128, 1), 256, 0, stream>>>(F23, W3f, b3, H3, K23, 16, 10);

    // log_softmax
    logsoftmax10<<<(BATCH + 255) / 256, 256, 0, stream>>>(H3, out);
}
